// DCRNN_Model_8581344657589
// MI455X (gfx1250) — matrix operand order repaired and checked by reading, not compiled, not run on hardware
//
#include <hip/hip_runtime.h>
#include <hip/hip_bf16.h>
#include <stdint.h>

// ---------------------------------------------------------------------------
// DCRNN (2 diffusion-GRU cells, zero initial state) on gfx1250.
// Hidden-state paths are algebraically dead (H0 = 0), so each cell is:
//   h = relu((1 - sigmoid(Dz)) * tanh(Dh)),
//   D* = X@(W[0,0]+W[1,0]) + Txo@W[0,1] + Txi@W[1,1] + b
// GEMMs run on v_wmma_f32_16x16x32_f16 (wave32, 16x16 tile per wave),
// graph diffusion via float global atomics, weights staged in LDS as f16.
// ---------------------------------------------------------------------------

#define N_NODES 50000
#define N_EDGES 800000
#define F_INF   32
#define H1F     50
#define H2F     20
#define ROWBLKS 3125   // 50000 / 16 exactly

typedef __attribute__((ext_vector_type(16))) _Float16 v16h;
typedef __attribute__((ext_vector_type(8)))  float    v8f;

// workspace layout (float units)
#define OFF_DEG_O 0
#define OFF_DEG_I (OFF_DEG_O + N_NODES)
#define OFF_TXO1  (OFF_DEG_I + N_NODES)            // N*32
#define OFF_TXI1  (OFF_TXO1 + N_NODES * 32)        // N*32
#define OFF_TXO2  (OFF_TXI1 + N_NODES * 32)        // N*64 (stride-64 rows, cols>=50 stay 0)
#define OFF_TXI2  (OFF_TXO2 + N_NODES * 64)        // N*64
#define ZERO_END  (OFF_TXI2 + N_NODES * 64)        // everything above needs zeroing (atomics)
#define OFF_H1    ZERO_END                         // N*64 (fully overwritten; padding cols == 0)
#define OFF_H2    (OFF_H1 + N_NODES * 64)          // N*20 (fully overwritten)
#define OFF_WNO   (OFF_H2 + N_NODES * 20)          // E
#define OFF_WNI   (OFF_WNO + N_EDGES)              // E
#define OFF_W16   (OFF_WNI + N_EDGES)              // f16 area: 4 x 6144 halfs
// f16-unit offsets inside W16 area:
//  [0)      W1z^T [64 cols][96 K]
//  [6144)   W1h^T [64 cols][96 K]
//  [12288)  W2z^T [32 cols][192 K]
//  [18432)  W2h^T [32 cols][192 K]

__global__ void k_zero(float* __restrict__ ws, int n4) {
    int i = blockIdx.x * blockDim.x + threadIdx.x;
    if (i < n4) ((float4*)ws)[i] = make_float4(0.f, 0.f, 0.f, 0.f);
}

__global__ void k_deg(const int* __restrict__ src, const int* __restrict__ dst,
                      const float* __restrict__ w,
                      float* __restrict__ dego, float* __restrict__ degi) {
    int e = blockIdx.x * blockDim.x + threadIdx.x;
    if (e < N_EDGES) {
        float we = w[e];
        atomicAdd(&dego[src[e]], we);
        atomicAdd(&degi[dst[e]], we);
    }
}

__global__ void k_norm(const int* __restrict__ src, const int* __restrict__ dst,
                       const float* __restrict__ w,
                       const float* __restrict__ dego, const float* __restrict__ degi,
                       float* __restrict__ wno, float* __restrict__ wni) {
    int e = blockIdx.x * blockDim.x + threadIdx.x;
    if (e < N_EDGES) {
        float we = w[e];
        float doo = dego[src[e]];
        float dii = degi[dst[e]];
        wno[e] = doo > 0.f ? we / doo : 0.f;
        wni[e] = dii > 0.f ? we / dii : 0.f;
    }
}

// Build combined, transposed, zero-padded f16 weights:
//  cell1: Wc[c][k], c<64 (50 used), k<96 : seg0 = W[0,0]+W[1,0], seg1 = W[0,1], seg2 = W[1,1]
//  cell2: Wc[c][k], c<32 (20 used), k<192 (3 segs of 64, rows 50..63 zero)
__global__ void k_prepw(const float* __restrict__ W1z, const float* __restrict__ W1h,
                        const float* __restrict__ W2z, const float* __restrict__ W2h,
                        _Float16* __restrict__ w16) {
    int t = blockIdx.x * blockDim.x + threadIdx.x;
    if (t < 12288) {                       // cell 1: 2 gates x 64 x 96
        int gate = t / 6144, rem = t % 6144;
        int c = rem / 96, k = rem % 96;
        const float* W = gate ? W1h : W1z;     // (2,2,82,50) flat: ((d*2+kk)*82+r)*50+c
        int seg = k >> 5, r = k & 31;
        float v = 0.f;
        if (c < H1F) {
            if (seg == 0)      v = W[(0 * 82 + r) * 50 + c] + W[(2 * 82 + r) * 50 + c];
            else if (seg == 1) v = W[(1 * 82 + r) * 50 + c];
            else               v = W[(3 * 82 + r) * 50 + c];
        }
        w16[gate * 6144 + c * 96 + k] = (_Float16)v;
    } else if (t < 24576) {                // cell 2: 2 gates x 32 x 192
        int off = t - 12288;
        int gate = off / 6144, rem = off % 6144;
        int c = rem / 192, k = rem % 192;
        const float* W = gate ? W2h : W2z;     // (2,2,70,20) flat: ((d*2+kk)*70+r)*20+c
        int seg = k / 64, r = k % 64;
        float v = 0.f;
        if (c < H2F && r < H1F) {
            if (seg == 0)      v = W[(0 * 70 + r) * 20 + c] + W[(2 * 70 + r) * 20 + c];
            else if (seg == 1) v = W[(1 * 70 + r) * 20 + c];
            else               v = W[(3 * 70 + r) * 20 + c];
        }
        w16[12288 + gate * 6144 + c * 192 + k] = (_Float16)v;
    }
}

// Txo[src] += wno * X[dst];  Txi[dst] += wni * X[src]   (32 features, stride 32)
__global__ void k_scatter1(const int* __restrict__ src, const int* __restrict__ dst,
                           const float* __restrict__ wno, const float* __restrict__ wni,
                           const float* __restrict__ x,
                           float* __restrict__ txo, float* __restrict__ txi) {
    int idx = blockIdx.x * blockDim.x + threadIdx.x;   // < E*32 = 25.6M
    int e = idx >> 5, f = idx & 31;
    if (e < N_EDGES) {
        int s = src[e], d = dst[e];
        atomicAdd(&txo[s * 32 + f], wno[e] * x[d * 32 + f]);
        atomicAdd(&txi[d * 32 + f], wni[e] * x[s * 32 + f]);
    }
}

// same on h1 (50 features, stride-64 rows)
__global__ void k_scatter2(const int* __restrict__ src, const int* __restrict__ dst,
                           const float* __restrict__ wno, const float* __restrict__ wni,
                           const float* __restrict__ h1,
                           float* __restrict__ txo, float* __restrict__ txi) {
    int idx = blockIdx.x * blockDim.x + threadIdx.x;   // < E*64 = 51.2M
    int e = idx >> 6, f = idx & 63;
    if (e < N_EDGES && f < H1F) {
        int s = src[e], d = dst[e];
        atomicAdd(&txo[s * 64 + f], wno[e] * h1[d * 64 + f]);
        atomicAdd(&txi[d * 64 + f], wni[e] * h1[s * 64 + f]);
    }
}

__device__ __forceinline__ float sigmf(float v) { return 1.f / (1.f + __expf(-v)); }

// Cell 1: per wave one 16x16 output tile, K = 3 segments x 32 -> 3 WMMA per gate.
// 8 waves/block: wave -> (ntile = w&3, rowblock = blk*2 + (w>>2)).
__global__ void __launch_bounds__(256) k_cell1(
        const float* __restrict__ x, const float* __restrict__ txo,
        const float* __restrict__ txi, const _Float16* __restrict__ w16,
        const float* __restrict__ bz, const float* __restrict__ bh,
        float* __restrict__ h1out) {
    __shared__ __align__(32) _Float16 sW[2][64 * 96];
    {
        const uint32_t* g = (const uint32_t*)w16;      // W1z^T then W1h^T = 6144 dwords
        uint32_t* sm = (uint32_t*)&sW[0][0];
        for (int i = threadIdx.x; i < 6144; i += 256) sm[i] = g[i];
    }
    __syncthreads();

    int wave = threadIdx.x >> 5, lane = threadIdx.x & 31;
    int nl = lane & 15, hh = lane >> 4;
    int ntile = wave & 3;
    int rowblock = blockIdx.x * 2 + (wave >> 2);
    if (rowblock >= ROWBLKS) return;                   // wave-uniform: EXEC stays full
    int rowbase = rowblock * 16;
    int arow = rowbase + nl;                           // A row handled by this lane
    int col = ntile * 16 + nl;                         // output column of this lane
    int kb = hh * 8;

    v8f accz = {}; v8f acch = {};
    const float* seg0 = x   + (size_t)arow * 32;
    const float* seg1 = txo + (size_t)arow * 32;
    const float* seg2 = txi + (size_t)arow * 32;
    const float* segp[3] = { seg0, seg1, seg2 };

#pragma unroll
    for (int s = 0; s < 3; ++s) {
        const float* ap = segp[s];
        v16h a;
#pragma unroll
        for (int j = 0; j < 8; ++j) a[j] = (_Float16)ap[2 * kb + j];            // K = 16h + j
#pragma unroll
        for (int j = 0; j < 8; ++j) a[8 + j] = (_Float16)ap[2 * kb + 8 + j];   // K = 16h + 8 + j
        v16h bzf = *(const v16h*)&sW[0][col * 96 + s * 32 + hh * 16];       // B[K][col]
        v16h bhf = *(const v16h*)&sW[1][col * 96 + s * 32 + hh * 16];
        accz = __builtin_amdgcn_wmma_f32_16x16x32_f16(false, a, false, bzf, (short)0, accz, false, false);
        acch = __builtin_amdgcn_wmma_f32_16x16x32_f16(false, a, false, bhf, (short)0, acch, false, false);
    }

    float biasz = (col < H1F) ? bz[col] : 0.f;
    float biash = (col < H1F) ? bh[col] : 0.f;
#pragma unroll
    for (int r = 0; r < 8; ++r) {
        int row = rowbase + r + 8 * hh;
        float z  = sigmf(accz[r] + biasz);
        float ht = tanhf(acch[r] + biash);
        float v  = (1.f - z) * ht;
        h1out[(size_t)row * 64 + col] = v > 0.f ? v : 0.f;  // cols>=50 compute exactly 0
    }
}

// Cell 2: K = 3 segments x 64 (rows 50..63 read h1/Tx padding zeros) -> 6 WMMA per gate.
// 8 waves/block: wave -> (ntile = w&1, rowblock = blk*4 + (w>>1)).
__global__ void __launch_bounds__(256) k_cell2(
        const float* __restrict__ h1, const float* __restrict__ txo,
        const float* __restrict__ txi, const _Float16* __restrict__ w16,
        const float* __restrict__ bz, const float* __restrict__ bh,
        float* __restrict__ h2out) {
    __shared__ __align__(32) _Float16 sW[2][32 * 192];
    {
        const uint32_t* g = ((const uint32_t*)w16) + 6144;  // W2z^T then W2h^T
        uint32_t* sm = (uint32_t*)&sW[0][0];
        for (int i = threadIdx.x; i < 6144; i += 256) sm[i] = g[i];
    }
    __syncthreads();

    int wave = threadIdx.x >> 5, lane = threadIdx.x & 31;
    int nl = lane & 15, hh = lane >> 4;
    int ntile = wave & 1;
    int rowblock = blockIdx.x * 4 + (wave >> 1);
    if (rowblock >= ROWBLKS) return;
    int rowbase = rowblock * 16;
    int arow = rowbase + nl;
    int col = ntile * 16 + nl;
    int kb = hh * 8;

    v8f accz = {}; v8f acch = {};
    const float* seg0 = h1  + (size_t)arow * 64;
    const float* seg1 = txo + (size_t)arow * 64;
    const float* seg2 = txi + (size_t)arow * 64;
    const float* segp[3] = { seg0, seg1, seg2 };

#pragma unroll
    for (int s = 0; s < 6; ++s) {
        const float* ap = segp[s >> 1] + (s & 1) * 32;
        v16h a;
#pragma unroll
        for (int j = 0; j < 8; ++j) a[j] = (_Float16)ap[2 * kb + j];
#pragma unroll
        for (int j = 0; j < 8; ++j) a[8 + j] = (_Float16)ap[2 * kb + 8 + j];
        v16h bzf = *(const v16h*)&sW[0][col * 192 + s * 32 + hh * 16];
        v16h bhf = *(const v16h*)&sW[1][col * 192 + s * 32 + hh * 16];
        accz = __builtin_amdgcn_wmma_f32_16x16x32_f16(false, a, false, bzf, (short)0, accz, false, false);
        acch = __builtin_amdgcn_wmma_f32_16x16x32_f16(false, a, false, bhf, (short)0, acch, false, false);
    }

    float biasz = (col < H2F) ? bz[col] : 0.f;
    float biash = (col < H2F) ? bh[col] : 0.f;
#pragma unroll
    for (int r = 0; r < 8; ++r) {
        int row = rowbase + r + 8 * hh;
        float z  = sigmf(accz[r] + biasz);
        float ht = tanhf(acch[r] + biash);
        float v  = (1.f - z) * ht;
        if (col < H2F) h2out[(size_t)row * 20 + col] = v > 0.f ? v : 0.f;
    }
}

__global__ void k_final(const float* __restrict__ h2, const float* __restrict__ lw,
                        const float* __restrict__ lb, float* __restrict__ out) {
    int n = blockIdx.x * blockDim.x + threadIdx.x;
    if (n < N_NODES) {
        float acc = lb[0];
#pragma unroll
        for (int j = 0; j < H2F; ++j) acc += h2[(size_t)n * 20 + j] * lw[j];
        out[n] = acc;
    }
}

extern "C" void kernel_launch(void* const* d_in, const int* in_sizes, int n_in,
                              void* d_out, int out_size, void* d_ws, size_t ws_size,
                              hipStream_t stream) {
    // Inputs assumed in pytree (sorted-key) flatten order of setup_inputs():
    //  0:x 1:edge_index 2:edge_weight
    //  3:d1.W_h 4:d1.W_r 5:d1.W_z 6:d1.b_h 7:d1.b_r 8:d1.b_z
    //  9:d2.W_h 10:d2.W_r 11:d2.W_z 12:d2.b_h 13:d2.b_r 14:d2.b_z
    //  15:lin_b 16:lin_w        (W_r / b_r are dead: H0 == 0)
    const float* x    = (const float*)d_in[0];
    const int*   ei   = (const int*)d_in[1];
    const float* ew   = (const float*)d_in[2];
    const float* W1h  = (const float*)d_in[3];
    const float* W1z  = (const float*)d_in[5];
    const float* b1h  = (const float*)d_in[6];
    const float* b1z  = (const float*)d_in[8];
    const float* W2h  = (const float*)d_in[9];
    const float* W2z  = (const float*)d_in[11];
    const float* b2h  = (const float*)d_in[12];
    const float* b2z  = (const float*)d_in[14];
    const float* linb = (const float*)d_in[15];
    const float* linw = (const float*)d_in[16];

    float* ws = (float*)d_ws;
    const int* src = ei;
    const int* dst = ei + N_EDGES;
    float* dego = ws + OFF_DEG_O;
    float* degi = ws + OFF_DEG_I;
    float* txo1 = ws + OFF_TXO1;
    float* txi1 = ws + OFF_TXI1;
    float* txo2 = ws + OFF_TXO2;
    float* txi2 = ws + OFF_TXI2;
    float* h1   = ws + OFF_H1;
    float* h2   = ws + OFF_H2;
    float* wno  = ws + OFF_WNO;
    float* wni  = ws + OFF_WNI;
    _Float16* w16 = (_Float16*)(ws + OFF_W16);

    const int zero4 = ZERO_END / 4;  // 9,700,000 floats -> 2,425,000 float4
    k_zero<<<(zero4 + 255) / 256, 256, 0, stream>>>(ws, zero4);
    k_deg<<<(N_EDGES + 255) / 256, 256, 0, stream>>>(src, dst, ew, dego, degi);
    k_norm<<<(N_EDGES + 255) / 256, 256, 0, stream>>>(src, dst, ew, dego, degi, wno, wni);
    k_prepw<<<96, 256, 0, stream>>>(W1z, W1h, W2z, W2h, w16);
    k_scatter1<<<(N_EDGES * 32) / 256, 256, 0, stream>>>(src, dst, wno, wni, x, txo1, txi1);
    k_cell1<<<(ROWBLKS + 1) / 2, 256, 0, stream>>>(x, txo1, txi1, w16, b1z, b1h, h1);
    k_scatter2<<<(N_EDGES * 64) / 256, 256, 0, stream>>>(src, dst, wno, wni, h1, txo2, txi2);
    k_cell2<<<(ROWBLKS + 3) / 4, 256, 0, stream>>>(h1, txo2, txi2, w16, b2z, b2h, h2);
    k_final<<<(N_NODES + 255) / 256, 256, 0, stream>>>(h2, linw, linb, (float*)d_out);
}
